// DERNN_59545426592302
// MI455X (gfx1250) — matrix operand order repaired and checked by reading, not compiled, not run on hardware
//
#include <hip/hip_runtime.h>

// ---------------------------------------------------------------------------
// DERNN on full binary trees (B=512, N=127, H=256, E=300, Q=10, DEPTH=7)
// Matmuls (x@W.T: 30 GF, h@U.T: 25 GF) via v_wmma_f32_16x16x32_f16 (f32 acc).
// q@D.T collapses to a 10x768 table. Tree recurrence level-by-level with a
// fused elementwise combine (full tree => exactly 2 children per parent).
// Level GEMM A-operand staged HBM->LDS with global_load_async_to_lds_b128
// (ASYNCcnt) into a double-buffered LDS tile; B fragments preloaded in bulk
// so loadcnt drains overlap the WMMA issue.
// ---------------------------------------------------------------------------

typedef __attribute__((ext_vector_type(16))) _Float16 v16h;
typedef __attribute__((ext_vector_type(8)))  _Float16 v8h;
typedef __attribute__((ext_vector_type(8)))  float    v8f;

#define B_SZ   512
#define NNODE  127
#define NN     (B_SZ * NNODE)      // 65024
#define H_SZ   256
#define E_SZ   300
#define E_PAD  320
#define H3     768
#define LDSA   40                  // halves per A row in LDS (32 data + 8 pad)

__device__ __forceinline__ float fsig(float x)  { return 1.0f / (1.0f + __expf(-x)); }
__device__ __forceinline__ float ftanh(float x) { return 1.0f - 2.0f / (__expf(2.0f * x) + 1.0f); }

// CDNA5 async LDS copy: per-lane 16B, tracked by ASYNCcnt (ISA 15.18 op 98)
__device__ __forceinline__ void async_b128(void* lds_dst, const void* gsrc) {
    unsigned int loff = (unsigned int)(unsigned long long)lds_dst; // addr[31:0] = LDS offset
    asm volatile("global_load_async_to_lds_b128 %0, %1, off"
                 :: "v"(loff), "v"(gsrc) : "memory");
}
__device__ __forceinline__ void wait_async0() {
    asm volatile("s_wait_asynccnt 0x0" ::: "memory");
}

__device__ __forceinline__ v8h cvt8(float4 a, float4 b) {
    v8h r;
    r[0] = (_Float16)a.x; r[1] = (_Float16)a.y; r[2] = (_Float16)a.z; r[3] = (_Float16)a.w;
    r[4] = (_Float16)b.x; r[5] = (_Float16)b.y; r[6] = (_Float16)b.z; r[7] = (_Float16)b.w;
    return r;
}

// ---------------------------------------------------------------------------
// small prep kernels
// ---------------------------------------------------------------------------
__global__ void convert_f16(const float* __restrict__ src, _Float16* __restrict__ dst,
                            int rows, int scols, int dcols) {
    int i = blockIdx.x * blockDim.x + threadIdx.x;
    if (i >= rows * dcols) return;
    int r = i / dcols, c = i - r * dcols;
    dst[i] = (c < scols) ? (_Float16)src[r * scols + c] : (_Float16)0.0f;
}

// qD[qi][j] = dot(q[qi], D[j]) : 10 x 768 (tiny)
__global__ void qd_kernel(const float* __restrict__ q, const float* __restrict__ D,
                          float* __restrict__ qD) {
    int i = blockIdx.x * blockDim.x + threadIdx.x;
    if (i >= 10 * H3) return;
    int qi = i / H3, j = i - qi * H3;
    const float* qa = q + qi * H_SZ;
    const float* da = D + j * H_SZ;
    float s = 0.f;
    #pragma unroll 8
    for (int k = 0; k < H_SZ; ++k) s += qa[k] * da[k];
    qD[i] = s;
}

// ---------------------------------------------------------------------------
// WMMA GEMM:  C[M x 768] = Arows[M x K] @ B16^T  (B16 row-major [768 x ldb]
// f16 == exactly the [N][K] layout the WMMA B-fragment wants; L2-resident).
// Block tile 128x128, 8 waves (4m x 2n), wave tile 32x64 (2x4 C-frags).
// EMBED: A row m = idx2vec[tok[m]] (f32, K=300 padded to 320), sync staging.
// !EMBED: A row m -> tree node, A = hf16[node] (f16, K=256), async staging.
// Double-buffered LDS A tile; one barrier per K-chunk.
// ---------------------------------------------------------------------------
template<bool EMBED>
__launch_bounds__(256)
__global__ void gemm128x128(const float* __restrict__ Af32,
                            const _Float16* __restrict__ Af16,
                            const int* __restrict__ tok,
                            const _Float16* __restrict__ B16,
                            float* __restrict__ C,
                            int nk, int Ksrc, int ldb, int dc)
{
    __shared__ _Float16 As[2][128 * LDSA];

    const int tid  = threadIdx.x;
    const int lane = tid & 31;
    const int wave = tid >> 5;
    const int wm   = wave >> 1;                   // 0..3
    const int wn   = wave & 1;                    // 0..1
    const int nbase = blockIdx.y * 128 + wn * 64;

    // --- per-thread staging assignment: row = tid/2, 16-half segment ---
    const int srow = tid >> 1;                    // 0..127
    const int sseg = (tid & 1) * 16;              // 0 or 16
    const int grow = blockIdx.x * 128 + srow;     // global A row
    const float*    arow32 = nullptr;
    const _Float16* arow16 = nullptr;
    if (EMBED) {
        arow32 = Af32 + (long)tok[grow] * Ksrc + sseg;
    } else {
        const int c     = 1 << dc;
        const int batch = grow >> dc;
        const int i     = grow & (c - 1);
        const int node  = batch * NNODE + (c - 1) + i;
        arow16 = Af16 + (long)node * Ksrc + sseg;
    }
    _Float16* dstb[2] = { &As[0][srow * LDSA + sseg], &As[1][srow * LDSA + sseg] };

    auto stage = [&](int i) {
        const int k0 = i * 32;
        _Float16* dst = dstb[i & 1];
        if (EMBED) {
            const int kw = Ksrc - k0 - sseg;      // valid elems in this segment
            if (kw >= 16) {                       // full 16: pure b128 traffic
                const float4* s = (const float4*)(arow32 + k0);
                float4 f0 = s[0], f1 = s[1], f2 = s[2], f3 = s[3];
                *(v8h*)(dst)     = cvt8(f0, f1);
                *(v8h*)(dst + 8) = cvt8(f2, f3);
            } else {                              // single ragged tail chunk
                #pragma unroll
                for (int e = 0; e < 16; ++e)
                    dst[e] = (e < kw) ? (_Float16)arow32[k0 + e] : (_Float16)0.0f;
            }
        } else {                                  // f16 source: async HBM->LDS
            async_b128(dst,     arow16 + k0);
            async_b128(dst + 8, arow16 + k0 + 8);
        }
    };

    v8f zero;
    #pragma unroll
    for (int e = 0; e < 8; ++e) zero[e] = 0.0f;
    v8f acc[2][4];
    #pragma unroll
    for (int mt = 0; mt < 2; ++mt)
        #pragma unroll
        for (int nt = 0; nt < 4; ++nt) acc[mt][nt] = zero;

    const int kh = lane >> 4;                     // K-half selector
    const int ml = lane & 15;

    stage(0);
    for (int i = 0; i < nk; ++i) {
        if (!EMBED) wait_async0();                // ASYNCcnt -> LDS data landed
        __syncthreads();                          // all waves staged / done reading
        if (i + 1 < nk) stage(i + 1);             // overlap next chunk

        const _Float16* Ab = As[i & 1];
        const int k0 = i * 32;

        // A fragments (16x32 f16): lane ml holds K {kh*16..+15}
        v16h afrag[2];
        #pragma unroll
        for (int mt = 0; mt < 2; ++mt) {
            const _Float16* ap = &Ab[(wm * 32 + mt * 16 + ml) * LDSA];
            v8h lo = *(const v8h*)(ap + kh * 16);
            v8h hi = *(const v8h*)(ap + kh * 16 + 8);
            afrag[mt] = __builtin_shufflevector(lo, hi,
                0, 1, 2, 3, 4, 5, 6, 7, 8, 9, 10, 11, 12, 13, 14, 15);
        }

        // Preload ALL B fragments (distinct regs => one load clause, and the
        // loadcnt drains overlap earlier WMMA issue instead of gating them).
        const _Float16* Bp = B16 + (long)(nbase + ml) * ldb + k0 + kh * 16;
        v16h bfrag[4];
        #pragma unroll
        for (int nt = 0; nt < 4; ++nt)
            bfrag[nt] = *(const v16h*)(Bp + (long)(nt * 16) * ldb);

        #pragma unroll
        for (int nt = 0; nt < 4; ++nt)
            #pragma unroll
            for (int mt = 0; mt < 2; ++mt)
                acc[mt][nt] = __builtin_amdgcn_wmma_f32_16x16x32_f16(
                    false, afrag[mt], false, bfrag[nt], (short)0, acc[mt][nt],
                    false, false);
    }

    // store C: frag element e -> M = e + (lane>=16 ? 8 : 0), N = lane&15
    const int mrow = blockIdx.x * 128 + wm * 32 + kh * 8;
    #pragma unroll
    for (int mt = 0; mt < 2; ++mt)
        #pragma unroll
        for (int nt = 0; nt < 4; ++nt) {
            float* cp = C + (long)(mrow + mt * 16) * H3 + nbase + nt * 16 + ml;
            #pragma unroll
            for (int e = 0; e < 8; ++e) cp[(long)e * H3] = acc[mt][nt][e];
        }
}

// ---------------------------------------------------------------------------
// leaves (depth 6, tree idx 63..126):
// h = tanh( sigmoid(xWiu_i + q9Diu_i + b_i) * tanh(xWiu_u + q9Diu_u + b_u) )
// ---------------------------------------------------------------------------
__global__ void leaf_kernel(const float* __restrict__ xW3, const float* __restrict__ qD,
                            const float* __restrict__ b, _Float16* __restrict__ hf16) {
    long i = (long)blockIdx.x * blockDim.x + threadIdx.x;   // 512*64*256 elems
    if (i >= (long)B_SZ * 64 * H_SZ) return;
    const int  j     = (int)(i & 255);
    const long li    = i >> 8;
    const int  batch = (int)(li >> 6);
    const int  leaf  = (int)(li & 63);
    const int  node  = batch * NNODE + 63 + leaf;
    const float* xr  = xW3 + (long)node * H3;
    const float* q9  = qD + 9 * H3;
    const float zi = xr[256 + j] + q9[256 + j] + b[256 + j];
    const float zu = xr[512 + j] + q9[512 + j] + b[512 + j];
    hf16[(long)node * H_SZ + j] = (_Float16)ftanh(fsig(zi) * ftanh(zu));
}

// ---------------------------------------------------------------------------
// per-level combine: parent p (depth dc-1) from children (depth dc).
// Y = h_children @ U.T with rows r = batch*2^dc + (child_tree_idx - (2^dc-1)).
// ---------------------------------------------------------------------------
__global__ void combine_kernel(const float* __restrict__ xW3, const float* __restrict__ Y,
                               const float* __restrict__ qD, const float* __restrict__ b,
                               const int* __restrict__ dep, _Float16* __restrict__ hf16,
                               int dc, float* __restrict__ out) {
    const int cp = 1 << (dc - 1);                 // parents per batch
    const long total = (long)B_SZ * cp * H_SZ;
    long i = (long)blockIdx.x * blockDim.x + threadIdx.x;
    if (i >= total) return;
    const int  j     = (int)(i & 255);
    const long pi    = i >> 8;
    const int  batch = (int)(pi >> (dc - 1));
    const int  ip    = (int)(pi & (cp - 1));
    const int  pt    = (cp - 1) + ip;             // parent tree index
    const int  pnode = batch * NNODE + pt;
    const int  c1    = batch * NNODE + 2 * pt + 1;
    const int  c2    = c1 + 1;
    const long r1    = (long)batch * (1 << dc) + 2 * ip;
    const long r2    = r1 + 1;

    const float* xr  = xW3 + (long)pnode * H3;
    const float* y1  = Y + r1 * H3;
    const float* y2  = Y + r2 * H3;
    const float* qd1 = qD + (long)dep[c1] * H3;
    const float* qd2 = qD + (long)dep[c2] * H3;

    const float f1 = fsig(xr[j] + y1[j] + qd1[j] + b[j]);
    const float f2 = fsig(xr[j] + y2[j] + qd2[j] + b[j]);
    const float fsum = f1 * (float)hf16[(long)c1 * H_SZ + j]
                     + f2 * (float)hf16[(long)c2 * H_SZ + j];

    const float zi = xr[256 + j] + y1[256 + j] + y2[256 + j] + qd1[256 + j] + qd2[256 + j] + b[256 + j];
    const float zu = xr[512 + j] + y1[512 + j] + y2[512 + j] + qd1[512 + j] + qd2[512 + j] + b[512 + j];
    const float hv = ftanh(fsig(zi) * ftanh(zu) + fsum);

    hf16[(long)pnode * H_SZ + j] = (_Float16)hv;
    if (out) out[batch * H_SZ + j] = hv;          // dc==1: parent is the root
}

// ---------------------------------------------------------------------------
extern "C" void kernel_launch(void* const* d_in, const int* in_sizes, int n_in,
                              void* d_out, int out_size, void* d_ws, size_t ws_size,
                              hipStream_t stream)
{
    const int*   tokens  = (const int*)  d_in[0];
    const int*   dep     = (const int*)  d_in[1];
    const float* idx2vec = (const float*)d_in[2];
    const float* q       = (const float*)d_in[3];
    const float* W       = (const float*)d_in[4];
    const float* U       = (const float*)d_in[5];
    const float* D       = (const float*)d_in[6];
    const float* b       = (const float*)d_in[7];
    float*       out     = (float*)d_out;
    (void)in_sizes; (void)n_in; (void)out_size; (void)ws_size;

    char* ws = (char*)d_ws;
    size_t o = 0;
    _Float16* W16  = (_Float16*)(ws + o); o += (size_t)H3 * E_PAD * 2;  // 768x320 f16
    _Float16* U16  = (_Float16*)(ws + o); o += (size_t)H3 * H_SZ * 2;   // 768x256 f16
    float*    qD   = (float*)   (ws + o); o += (size_t)10 * H3 * 4;     // 10x768
    o = (o + 255) & ~(size_t)255;
    float*    xW3  = (float*)   (ws + o); o += (size_t)NN * H3 * 4;     // 65024x768
    _Float16* hf16 = (_Float16*)(ws + o); o += (size_t)NN * H_SZ * 2;   // 65024x256 f16
    o = (o + 255) & ~(size_t)255;
    float*    Y    = (float*)   (ws + o); o += (size_t)32768 * H3 * 4;  // max level

    // prep: f16 weights + q@D.T table
    convert_f16<<<(H3 * E_PAD + 255) / 256, 256, 0, stream>>>(W, W16, H3, E_SZ, E_PAD);
    convert_f16<<<(H3 * H_SZ + 255) / 256, 256, 0, stream>>>(U, U16, H3, H_SZ, H_SZ);
    qd_kernel<<<(10 * H3 + 255) / 256, 256, 0, stream>>>(q, D, qD);

    // xW3 = gather(idx2vec, tokens) @ W.T   (M=65024, N=768, K=300->320)
    gemm128x128<true><<<dim3(NN / 128, H3 / 128), 256, 0, stream>>>(
        idx2vec, nullptr, tokens, W16, xW3, E_PAD / 32, E_SZ, E_PAD, 0);

    // leaf hidden states
    leaf_kernel<<<(B_SZ * 64 * H_SZ) / 256, 256, 0, stream>>>(xW3, qD, b, hf16);

    // bottom-up levels: dc = children depth (6 -> 1)
    for (int dc = 6; dc >= 1; --dc) {
        const int M = B_SZ << dc;                 // children rows this level
        gemm128x128<false><<<dim3(M / 128, H3 / 128), 256, 0, stream>>>(
            nullptr, hf16, nullptr, U16, Y, H_SZ / 32, H_SZ, H_SZ, dc);
        const long total = (long)B_SZ * (1 << (dc - 1)) * H_SZ;
        combine_kernel<<<(int)(total / 256), 256, 0, stream>>>(
            xW3, Y, qD, b, dep, hf16, dc, (dc == 1) ? out : nullptr);
    }
}
